// GraphReadInLSTM_34832184770750
// MI455X (gfx1250) — matrix operand order repaired and checked by reading, not compiled, not run on hardware
//
#include <hip/hip_runtime.h>
#include <hip/hip_bf16.h>

// ---------------------------------------------------------------------------
// LSTM rollout, MI455X (gfx1250).  batch=1, D=H=1024, N steps (device scalar).
// Weights live in VGPRs across all steps (preloaded bf16 A-tiles); per step:
// h -> LDS (one coalesced clause), ALL B chunks preloaded from LDS to regs
// (sched barrier keeps them hoisted), 8 back-to-back v_wmma_f32_16x16x32_bf16,
// fixed-order LDS tree reduce, distributed-flag device barrier with
// double-buffered h state.
// ---------------------------------------------------------------------------

typedef __attribute__((ext_vector_type(16))) __bf16 v16bf;
typedef __attribute__((ext_vector_type(8)))  float  v8f;

constexpr int HID = 1024;    // hidden dim
constexpr int G4  = 4096;    // 4*HID
constexpr int NWG = 64;      // persistent workgroups
constexpr int JPW = 16;      // hidden indices per WG (64*16 = 1024)
constexpr int NTH = 512;     // threads per WG = 16 waves
constexpr int FLAG_STRIDE = 16;  // dwords between per-WG flags (64 B lines)

union BF32B { uint4 q[2]; v16bf v; };

__device__ __forceinline__ void sched_fence() {
#if __has_builtin(__builtin_amdgcn_sched_barrier)
    __builtin_amdgcn_sched_barrier(0);
#else
    asm volatile("" ::: "memory");
#endif
}

// ---- one-time: cast W_hh fp32 -> bf16 (row-major [4096][1024]) -------------
__global__ void lstm_cast_whh(const float* __restrict__ W,
                              __hip_bfloat16* __restrict__ Wb, int n) {
    int i = blockIdx.x * blockDim.x + threadIdx.x;
    if (i < n) Wb[i] = __float2bfloat16(W[i]);
}

// ---- one-time: x_gates[4096] = W_ih @ x + b_ih + b_hh (fp32) ---------------
__global__ void lstm_xgates(const float* __restrict__ Wih,
                            const float* __restrict__ x,
                            const float* __restrict__ bih,
                            const float* __restrict__ bhh,
                            float* __restrict__ xg) {
    int r = blockIdx.x * blockDim.x + threadIdx.x;   // 0..4095
    const float4* wr = (const float4*)(Wih + (size_t)r * HID);
    const float4* xv = (const float4*)x;
    float s = 0.0f;
    for (int k = 0; k < HID / 4; ++k) {
        float4 a = wr[k], b = xv[k];
        s += a.x * b.x + a.y * b.y + a.z * b.z + a.w * b.w;
    }
    xg[r] = s + bih[r] + bhh[r];
}

// ---- one-time: zero both h buffers (bf16) and all step flags ---------------
__global__ void lstm_init(__hip_bfloat16* __restrict__ hb,
                          unsigned* __restrict__ flags) {
    int i = threadIdx.x;
    for (int j = i; j < 2 * HID; j += 256) hb[j] = __float2bfloat16(0.0f);
    for (int j = i; j < NWG * FLAG_STRIDE; j += 256) flags[j] = 0u;
}

// ---- persistent recurrent kernel -------------------------------------------
__global__ __launch_bounds__(NTH, 1)
void lstm_roll(const __hip_bfloat16* __restrict__ Wb,   // [4096][1024] bf16
               const float* __restrict__ xg,            // [4096] fp32
               __hip_bfloat16* __restrict__ hb,         // [2][1024] bf16 state
               unsigned* __restrict__ flags,            // [NWG] padded flags
               float* __restrict__ out,                 // [N][1024] fp32
               const int* __restrict__ dN) {
    __shared__ __align__(16) __hip_bfloat16 h_lds[HID]; // staged h (2 KB)
    __shared__ float part[256];                         // [gate][ksub][jl]
    __shared__ float xg_lds[64];

    const int N    = *dN;
    const int jb   = blockIdx.x * JPW;
    const int tid  = threadIdx.x;
    const int lane = tid & 31;
    const int wv   = tid >> 5;     // wave 0..15
    const int q    = wv >> 2;      // gate 0..3
    const int ks   = wv & 3;       // K quarter: [256*ks, 256*ks+256)
    const int m    = lane & 15;    // row within 16x16 tile
    const int gr   = lane >> 4;    // half-wave group

    // ---- preload this wave's A tiles (16 rows x 256 K) into registers ----
    const __hip_bfloat16* arow = Wb + (size_t)(q * HID + jb + m) * HID;
    BF32B Areg[8];
    #pragma unroll
    for (int kk = 0; kk < 8; ++kk) {
        const int k0 = ks * 256 + kk * 32;
        // A (16-bit 16x32): lanes 0-15 K={0..7,16..23}, lanes 16-31 K={8..15,24..31}
        Areg[kk].q[0] = *(const uint4*)(arow + k0 + 8 * gr);
        Areg[kk].q[1] = *(const uint4*)(arow + k0 + 16 + 8 * gr);
    }

    if (tid < 64)
        xg_lds[tid] = xg[(tid >> 4) * HID + jb + (tid & 15)];

    float cprev = 0.0f;            // cell state, owned by threads 0..15
    __syncthreads();

    for (int t = 0; t < N; ++t) {
        const __hip_bfloat16* hin  = hb + ((t & 1) ? HID : 0);
        __hip_bfloat16*       hout = hb + ((t & 1) ? 0 : HID);

        // ---- stage h into LDS: 512 threads x 1 dword = 2 KB ----
        ((unsigned*)h_lds)[tid] = ((const unsigned*)hin)[tid];
        __syncthreads();

        // ---- preload ALL B chunks (h replicated over N columns) from LDS ----
        BF32B Breg[8];
        #pragma unroll
        for (int kk = 0; kk < 8; ++kk) {
            const int base = ks * 256 + kk * 32 + 8 * gr;
            // B (32x16): half-wave group gr holds K = {8*gr..8*gr+7, 16+8*gr..16+8*gr+7}
            Breg[kk].q[0] = *(const uint4*)(h_lds + base);
            Breg[kk].q[1] = *(const uint4*)(h_lds + base + 16);
        }
        sched_fence();   // keep the 16 ds_load_b128 hoisted above the chain

        // ---- 8 back-to-back WMMAs, D->C accumulation ----
        v8f acc = {0.f, 0.f, 0.f, 0.f, 0.f, 0.f, 0.f, 0.f};
        #pragma unroll
        for (int kk = 0; kk < 8; ++kk)
            acc = __builtin_amdgcn_wmma_f32_16x16x32_bf16(
                false, Areg[kk].v, false, Breg[kk].v, (short)0, acc, false, false);

        // lanes 0 / 16 hold M = 0..7 / 8..15 in acc[0..7]
        if (m == 0) {
            #pragma unroll
            for (int r = 0; r < 8; ++r)
                part[(q * 4 + ks) * 16 + gr * 8 + r] = acc[r];
        }
        __syncthreads();

        // ---- elementwise LSTM cell (threads 0..15, one hidden index each) ----
        if (tid < 16) {
            float gs[4];
            #pragma unroll
            for (int q2 = 0; q2 < 4; ++q2)
                gs[q2] = (((part[(q2 * 4 + 0) * 16 + tid]  +
                            part[(q2 * 4 + 1) * 16 + tid]) +
                            part[(q2 * 4 + 2) * 16 + tid]) +
                            part[(q2 * 4 + 3) * 16 + tid]) +
                         xg_lds[q2 * 16 + tid];
            float iv = 1.0f / (1.0f + expf(-gs[0]));
            float fv = 1.0f / (1.0f + expf(-gs[1]));
            float gv = tanhf(gs[2]);
            float ov = 1.0f / (1.0f + expf(-gs[3]));
            float c  = fv * cprev + iv * gv;
            cprev = c;
            float h = ov * tanhf(c);
            out[(size_t)t * HID + jb + tid] = h;     // fp32 emission
            hout[jb + tid] = __float2bfloat16(h);    // state for next step
        }

        // ---- device-wide step barrier: distributed per-WG flags ----
        __threadfence();
        __syncthreads();
        if (tid == 0)
            __hip_atomic_store(&flags[(size_t)blockIdx.x * FLAG_STRIDE],
                               (unsigned)(t + 1), __ATOMIC_RELEASE,
                               __HIP_MEMORY_SCOPE_AGENT);
        if (tid < NWG)
            while (__hip_atomic_load(&flags[(size_t)tid * FLAG_STRIDE],
                                     __ATOMIC_ACQUIRE,
                                     __HIP_MEMORY_SCOPE_AGENT) < (unsigned)(t + 1))
                __builtin_amdgcn_s_sleep(1);
        __syncthreads();
        __threadfence();
    }
}

// ---------------------------------------------------------------------------
extern "C" void kernel_launch(void* const* d_in, const int* in_sizes, int n_in,
                              void* d_out, int out_size, void* d_ws, size_t ws_size,
                              hipStream_t stream) {
    const float* x    = (const float*)d_in[0];   // [1,1024]
    const float* Wih  = (const float*)d_in[1];   // [4096,1024]
    const float* Whh  = (const float*)d_in[2];   // [4096,1024]
    const float* bih  = (const float*)d_in[3];   // [4096]
    const float* bhh  = (const float*)d_in[4];   // [4096]
    const int*   dN   = (const int*)d_in[5];     // scalar num_nodes
    float*       out  = (float*)d_out;

    char* ws = (char*)d_ws;
    __hip_bfloat16* Wb  = (__hip_bfloat16*)ws;                              // 8 MB
    float*          xg  = (float*)(ws + (size_t)8 * 1024 * 1024);           // 16 KB
    __hip_bfloat16* hb  = (__hip_bfloat16*)(ws + (size_t)8 * 1024 * 1024 + 16384); // 4 KB
    unsigned*      flgs = (unsigned*)(ws + (size_t)8 * 1024 * 1024 + 16384 + 8192); // 4 KB

    const int nW = G4 * HID;                       // 4M elements
    lstm_cast_whh<<<(nW + 255) / 256, 256, 0, stream>>>(Whh, Wb, nW);
    lstm_xgates  <<<G4 / 256, 256, 0, stream>>>(Wih, x, bih, bhh, xg);
    lstm_init    <<<1, 256, 0, stream>>>(hb, flgs);
    lstm_roll    <<<NWG, NTH, 0, stream>>>(Wb, xg, hb, flgs, out, dN);
}
